// TGNMemory_64707977282177
// MI455X (gfx1250) — matrix operand order repaired and checked by reading, not compiled, not run on hardware
//
#include <hip/hip_runtime.h>

typedef unsigned short u16;
typedef unsigned int   u32;
typedef unsigned long long u64;

typedef __attribute__((ext_vector_type(16))) __bf16 v16bf;
typedef __attribute__((ext_vector_type(8)))  float  v8f;

#define M_CH   256
#define R_CH   128
#define T_CH   128
#define MSGD   768           // 2*M + R + T
#define GATES  768           // 3*M
#define MTILE  32            // nodes per block
#define BLKT   512           // 16 wave32 waves

__device__ __forceinline__ u16 f2bf(float f) {
    u32 u = __float_as_uint(f);
    u32 r = u + 0x7FFFu + ((u >> 16) & 1u);   // round-to-nearest-even
    return (u16)(r >> 16);
}

union Frag { v16bf v; uint4 u[2]; };

// ---------------------------------------------------------------- aggregation
__global__ void tgn_aggregate(const int* __restrict__ src, const int* __restrict__ dst,
                              const int* __restrict__ ts, u64* __restrict__ keys, int E) {
    int e2 = blockIdx.x * blockDim.x + threadIdx.x;
    if (e2 >= 2 * E) return;
    int e    = (e2 < E) ? e2 : e2 - E;
    int node = (e2 < E) ? src[e] : dst[e];
    u64 key = ((u64)(u32)ts[e] << 32) | (u64)(0xFFFFFFFFu - (u32)e2);
    atomicMax(keys + node, key);
}

// ---------------------------------------------------------- weight f32 -> bf16
__global__ void tgn_convert(const float* __restrict__ Wih, const float* __restrict__ Whh,
                            u16* __restrict__ WihB, u16* __restrict__ WhhB) {
    const int NIH = GATES * MSGD;       // 589824
    const int NHH = GATES * M_CH;       // 196608
    int i = blockIdx.x * blockDim.x + threadIdx.x;
    if (i < NIH)            WihB[i]       = f2bf(Wih[i]);
    else if (i < NIH + NHH) WhhB[i - NIH] = f2bf(Whh[i - NIH]);
}

// ------------------------------------------------------------------ fused GRU
__global__ void __launch_bounds__(BLKT)
tgn_gru(const int* __restrict__ src, const int* __restrict__ dst,
        const int* __restrict__ tsv, const float* __restrict__ raw,
        const float* __restrict__ memory, const int* __restrict__ last_update,
        const float* __restrict__ time_w, const float* __restrict__ time_b,
        const float* __restrict__ b_ih, const float* __restrict__ b_hh,
        const u16* __restrict__ WihB, const u16* __restrict__ WhhB,
        const u64* __restrict__ keys, float* __restrict__ out, int E) {

    extern __shared__ char dynsmem[];
    u16* Xs = (u16*)dynsmem;                 // [MTILE][768] bf16  (48 KB)
    u16* Hs = Xs + MTILE * MSGD;             // [MTILE][256] bf16  (16 KB)

    __shared__ int   sm_other[MTILE];
    __shared__ int   sm_e[MTILE];
    __shared__ float sm_dt[MTILE];
    __shared__ int   sm_has[MTILE];

    const int tid  = threadIdx.x;
    const int base = blockIdx.x * MTILE;

    // ---- step 0: decode winning event per node
    if (tid < MTILE) {
        int node = base + tid;
        u64 key  = keys[node];
        int has  = (key != 0ull);
        u32 pos  = 0xFFFFFFFFu - (u32)(key & 0xFFFFFFFFull);
        int e2   = has ? (int)pos : 0;
        int e    = (e2 < E) ? e2 : (e2 - E);
        sm_e[tid]     = e;
        sm_other[tid] = (e2 < E) ? dst[e] : src[e];
        sm_dt[tid]    = has ? ((float)tsv[e] - (float)last_update[node]) : 0.0f;
        sm_has[tid]   = has;
    }
    __syncthreads();

    // ---- step 1: build X (winning message, bf16) and H (memory, bf16) in LDS
    for (int idx = tid; idx < MTILE * 1024; idx += BLKT) {
        int i = idx >> 10, j = idx & 1023;
        int node = base + i;
        if (j < MSGD) {
            float v = 0.0f;
            if (sm_has[i]) {
                if (j < 256)       v = memory[(size_t)node * M_CH + j];
                else if (j < 512)  v = memory[(size_t)sm_other[i] * M_CH + (j - 256)];
                else if (j < 640)  v = raw[(size_t)sm_e[i] * R_CH + (j - 512)];
                else { int jt = j - 640; v = cosf(sm_dt[i] * time_w[jt] + time_b[jt]); }
            }
            Xs[i * MSGD + j] = f2bf(v);
        } else {
            int j2 = j - MSGD;
            Hs[i * M_CH + j2] = f2bf(memory[(size_t)node * M_CH + j2]);
        }
    }
    __syncthreads();

    // ---- step 2: WMMA GEMMs.  wave -> (mtile, 32-channel slice)
    const int wave = tid >> 5, lane = tid & 31;
    const int mt   = wave & 1;                 // which 16-node row tile
    const int ch   = (wave >> 1) * 32;         // channel slice [ch, ch+32)
    const int lm   = lane & 15, hi = lane >> 4;

    // acc[g*2+t] : g=0 r(gi+gh), g=1 z(gi+gh), g=2 i_n(gi), g=3 h_n(gh)
    v8f acc[8] = {};

    const u16* xrow = Xs + (mt * 16 + lm) * MSGD;
    for (int ks = 0; ks < MSGD / 32; ++ks) {
        Frag a;
        const uint4* pa = (const uint4*)(xrow + ks * 32 + hi * 8);
        a.u[0] = pa[0]; a.u[1] = pa[2];
#pragma unroll
        for (int g = 0; g < 3; ++g)
#pragma unroll
            for (int t = 0; t < 2; ++t) {
                int wrow = g * 256 + ch + t * 16 + lm;
                const uint4* pb = (const uint4*)(WihB + (size_t)wrow * MSGD + ks * 32 + hi * 8);
                Frag b; b.u[0] = pb[0]; b.u[1] = pb[2];
                acc[g * 2 + t] = __builtin_amdgcn_wmma_f32_16x16x32_bf16(
                    false, a.v, false, b.v, (short)0, acc[g * 2 + t], false, false);
            }
    }

    const u16* hrow = Hs + (mt * 16 + lm) * M_CH;
    for (int ks = 0; ks < M_CH / 32; ++ks) {
        Frag a;
        const uint4* pa = (const uint4*)(hrow + ks * 32 + hi * 8);
        a.u[0] = pa[0]; a.u[1] = pa[2];
#pragma unroll
        for (int gi = 0; gi < 3; ++gi) {       // gi=0->r, 1->z, 2->h_n
            int g     = (gi == 2) ? 3 : gi;
            int wbase = (gi == 2) ? 512 : gi * 256;
#pragma unroll
            for (int t = 0; t < 2; ++t) {
                int wrow = wbase + ch + t * 16 + lm;
                const uint4* pb = (const uint4*)(WhhB + (size_t)wrow * M_CH + ks * 32 + hi * 8);
                Frag b; b.u[0] = pb[0]; b.u[1] = pb[2];
                acc[g * 2 + t] = __builtin_amdgcn_wmma_f32_16x16x32_bf16(
                    false, a.v, false, b.v, (short)0, acc[g * 2 + t], false, false);
            }
        }
    }

    // ---- step 3: epilogue entirely in registers (C/D layout: lane<16 -> M=r, else M=r+8)
#pragma unroll
    for (int t = 0; t < 2; ++t) {
        int c = ch + t * 16 + lm;
        float bir = b_ih[c]       + b_hh[c];
        float biz = b_ih[256 + c] + b_hh[256 + c];
        float bin = b_ih[512 + c];
        float bhn = b_hh[512 + c];
#pragma unroll
        for (int r2 = 0; r2 < 8; ++r2) {
            int m    = mt * 16 + r2 + hi * 8;
            int node = base + m;
            float h  = memory[(size_t)node * M_CH + c];
            float gr  = acc[0 + t][r2];
            float gz  = acc[2 + t][r2];
            float gin = acc[4 + t][r2];
            float ghn = acc[6 + t][r2];
            float r = 1.0f / (1.0f + __expf(-(gr + bir)));
            float z = 1.0f / (1.0f + __expf(-(gz + biz)));
            float nn = tanhf(gin + bin + r * (ghn + bhn));
            float o  = sm_has[m] ? ((1.0f - z) * nn + z * h) : h;
            out[(size_t)node * M_CH + c] = o;
        }
    }
}

// ---------------------------------------------------------------------- host
extern "C" void kernel_launch(void* const* d_in, const int* in_sizes, int n_in,
                              void* d_out, int out_size, void* d_ws, size_t ws_size,
                              hipStream_t stream) {
    const int*   src  = (const int*)d_in[0];
    const int*   dst  = (const int*)d_in[1];
    const int*   tsv  = (const int*)d_in[2];
    const float* raw  = (const float*)d_in[3];
    const float* mem  = (const float*)d_in[4];
    const int*   lu   = (const int*)d_in[5];
    const float* tw   = (const float*)d_in[6];
    const float* tb   = (const float*)d_in[7];
    const float* Wih  = (const float*)d_in[8];
    const float* Whh  = (const float*)d_in[9];
    const float* bih  = (const float*)d_in[10];
    const float* bhh  = (const float*)d_in[11];
    float* out = (float*)d_out;

    const int E = in_sizes[0];
    const int N = in_sizes[4] / M_CH;

    u64* keys = (u64*)d_ws;
    u16* WihB = (u16*)((char*)d_ws + (size_t)N * 8);
    u16* WhhB = WihB + (size_t)GATES * MSGD;

    hipMemsetAsync(keys, 0, (size_t)N * 8, stream);

    int convN = GATES * MSGD + GATES * M_CH;
    tgn_convert<<<(convN + 255) / 256, 256, 0, stream>>>(Wih, Whh, WihB, WhhB);

    tgn_aggregate<<<(2 * E + 255) / 256, 256, 0, stream>>>(src, dst, tsv, keys, E);

    size_t lds = (size_t)MTILE * MSGD * 2 + (size_t)MTILE * M_CH * 2;   // 64 KB
    tgn_gru<<<(N + MTILE - 1) / MTILE, BLKT, lds, stream>>>(
        src, dst, tsv, raw, mem, lu, tw, tb, bih, bhh, WihB, WhhB, keys, out, E);
}
